// Head_68444598829701
// MI455X (gfx1250) — matrix operand order repaired and checked by reading, not compiled, not run on hardware
//
#include <hip/hip_runtime.h>
#include <hip/hip_bf16.h>

// ---------------------------------------------------------------------------
// Causal attention head for MI455X (gfx1250, wave32, WMMA).
//   x:[16,2048,1024]f32  Wk/Wq/Wv:[1024,128]f32  -> out:[16,2048,128]f32
// Stage 1: QKV projection GEMM in bf16 WMMA -> workspace (K stored transposed)
// Stage 2: flash-attention (online softmax) in bf16 WMMA, fp32 accumulate,
//          double-buffered K/V tiles staged with async global->LDS DMA.
// ---------------------------------------------------------------------------

#define B_ 16
#define T_ 2048
#define E_ 1024
#define H_ 128

typedef __attribute__((ext_vector_type(16))) __bf16 v16bf;
typedef __attribute__((ext_vector_type(8)))  float  v8f;
typedef int v4i __attribute__((vector_size(16)));   // matches builtin param type

union Frag16 {               // one 16-bit WMMA operand: 16 bf16 per lane = 32B
    v16bf v;
    uint4 u4[2];
};

static __device__ __forceinline__ __bf16 f2bf(float f) { return (__bf16)f; }

static __device__ __forceinline__ v8f wmma_bf16(v16bf a, v16bf b, v8f c) {
    return __builtin_amdgcn_wmma_f32_16x16x32_bf16(
        /*neg_a=*/false, a, /*neg_b=*/false, b,
        /*c_mod=*/(short)0, c, /*reuse_a=*/false, /*reuse_b=*/false);
}

// ---- async global->LDS DMA (gfx1250 GLOBAL_LOAD_ASYNC_TO_LDS_B128), guarded -
#if __has_builtin(__builtin_amdgcn_global_load_async_to_lds_b128)
#define HAS_ASYNC_LDS 1
#else
#define HAS_ASYNC_LDS 0
#endif

#if HAS_ASYNC_LDS
static __device__ __forceinline__ void async_copy16(void* lds_dst, const void* gsrc) {
    // AS casts via integers: generic==global addr; flat->LDS = low 32 bits
    __builtin_amdgcn_global_load_async_to_lds_b128(
        (__attribute__((address_space(1))) v4i*)(uintptr_t)gsrc,
        (__attribute__((address_space(3))) v4i*)(uint32_t)(uintptr_t)lds_dst,
        0, 0);
}
#endif

static __device__ __forceinline__ void wait_async0() {
#if __has_builtin(__builtin_amdgcn_s_wait_asynccnt)
    __builtin_amdgcn_s_wait_asynccnt(0);
#elif defined(__gfx1250__)
    asm volatile("s_wait_asynccnt 0" ::: "memory");
#endif
}

// ---------------------------------------------------------------------------
// Stage 1: out = x @ W for one of {Q,K,V} selected by blockIdx.y.
// Block: 128 threads (4 waves), tile M=64, N=128, K-chunk=32.
// Q,V written [B,T,H] bf16; K written transposed [B,H,T] bf16.
// ---------------------------------------------------------------------------
__global__ __launch_bounds__(128)
void qkv_proj_kernel(const float* __restrict__ x,
                     const float* __restrict__ Wq,
                     const float* __restrict__ Wk,
                     const float* __restrict__ Wv,
                     __bf16* __restrict__ Qb,
                     __bf16* __restrict__ Ktb,
                     __bf16* __restrict__ Vb)
{
    const int mat  = blockIdx.y;                // 0=Q, 1=K, 2=V
    const float* W = (mat == 0) ? Wq : ((mat == 1) ? Wk : Wv);

    const int tid  = threadIdx.x;
    const int w    = tid >> 5;                  // wave 0..3
    const int lane = tid & 31;
    const int hf   = lane >> 4;                 // lane half (A-frag K-half select)
    const int l16  = lane & 15;

    const int m0 = blockIdx.x * 64;             // 64 token rows per block

    __shared__ __bf16 xbuf[64][40];             // 64 x 32 (+8 pad), bf16
    __shared__ __bf16 wbuf[32][136];            // 32 x 128 (+8 pad), bf16

    v8f acc[8];
#pragma unroll
    for (int i = 0; i < 8; ++i) acc[i] = (v8f){};

    for (int kc = 0; kc < E_ / 32; ++kc) {
        const int ec = kc * 32;
        __syncthreads();                        // previous compute done reading LDS

        // ---- load x chunk [64 x 32] f32 -> bf16 LDS (512 float4, 4/thread)
#pragma unroll
        for (int j = 0; j < 4; ++j) {
            const int fid = tid + 128 * j;      // 0..511
            const int row = fid >> 3;           // 8 float4 per row
            const int c4  = (fid & 7) * 4;
            float4 xv = *(const float4*)&x[(size_t)(m0 + row) * E_ + ec + c4];
            xbuf[row][c4 + 0] = f2bf(xv.x);
            xbuf[row][c4 + 1] = f2bf(xv.y);
            xbuf[row][c4 + 2] = f2bf(xv.z);
            xbuf[row][c4 + 3] = f2bf(xv.w);
        }
        // ---- load W chunk [32 x 128] f32 -> bf16 LDS (1024 float4, 8/thread)
#pragma unroll
        for (int j = 0; j < 8; ++j) {
            const int fid = tid + 128 * j;      // 0..1023
            const int row = fid >> 5;           // 32 float4 per row
            const int c4  = (fid & 31) * 4;
            float4 wv = *(const float4*)&W[(size_t)(ec + row) * H_ + c4];
            wbuf[row][c4 + 0] = f2bf(wv.x);
            wbuf[row][c4 + 1] = f2bf(wv.y);
            wbuf[row][c4 + 2] = f2bf(wv.z);
            wbuf[row][c4 + 3] = f2bf(wv.w);
        }
        __syncthreads();

        // ---- A fragment (16x32): lane<16 holds K 0-7 & 16-23, lane>=16 K 8-15 & 24-31
        Frag16 a;
        {
            const __bf16* ar = &xbuf[w * 16 + l16][hf * 8];
            a.u4[0] = *(const uint4*)(ar);
            a.u4[1] = *(const uint4*)(ar + 16);
        }
        // ---- 8 N-tiles of B (32x16) and WMMA
#pragma unroll
        for (int hh = 0; hh < 8; ++hh) {
            Frag16 bfr;
#pragma unroll
            for (int s = 0; s < 16; ++s)
                bfr.v[s] = wbuf[(s & 7) + 8 * hf + 16 * (s >> 3)][hh * 16 + l16];
            acc[hh] = wmma_bf16(a.v, bfr.v, acc[hh]);
        }
    }

    // ---- epilogue: C/D layout -> element (VGPR r, lane) = row r+8*hf, col l16
#pragma unroll
    for (int hh = 0; hh < 8; ++hh) {
#pragma unroll
        for (int r = 0; r < 8; ++r) {
            const int m = m0 + w * 16 + r + 8 * hf;   // global token index
            const int h = hh * 16 + l16;
            const int bb = m >> 11;                   // / T_
            const int tt = m & (T_ - 1);
            const __bf16 val = f2bf(acc[hh][r]);
            if (mat == 0)      Qb [((size_t)bb * T_ + tt) * H_ + h] = val;
            else if (mat == 1) Ktb[((size_t)bb * H_ + h) * T_ + tt] = val;
            else               Vb [((size_t)bb * T_ + tt) * H_ + h] = val;
        }
    }
}

// ---------------------------------------------------------------------------
// Stage 2: flash attention. Block = 128 threads (4 waves), 64 queries/block,
// each wave owns 16 query rows; key tiles of 32, bf16 WMMA, online softmax.
// K/V tiles double-buffered in LDS; staged by async DMA when available.
// ---------------------------------------------------------------------------
__global__ __launch_bounds__(128)
void flash_attn_kernel(const __bf16* __restrict__ Qb,
                       const __bf16* __restrict__ Ktb,
                       const __bf16* __restrict__ Vb,
                       float* __restrict__ out)
{
    const int b    = blockIdx.y;
    const int qt   = blockIdx.x;                // 0..31 query tiles of 64
    const int tid  = threadIdx.x;
    const int w    = tid >> 5;
    const int lane = tid & 31;
    const int hf   = lane >> 4;
    const int l16  = lane & 15;
    const int qBase = qt * 64 + w * 16;

    __shared__ __bf16 kbuf[2][128][40];         // K^T slice: [h=128][keys=32] (+pad)
    __shared__ __bf16 vbuf[2][32][136];         // V slice:   [keys=32][h=128] (+pad)
    __shared__ __bf16 pbuf[4][16][32];          // per-wave P tile re-layout

    // ---- Q A-fragments in registers: 16 rows x 128 h = 4 fragments of 16x32
    v16bf aQ[4];
    {
        const __bf16* qrow = Qb + ((size_t)b * T_ + (qBase + l16)) * H_;
#pragma unroll
        for (int kk = 0; kk < 4; ++kk) {
            const int h0 = kk * 32 + hf * 8;
            Frag16 t;
            t.u4[0] = *(const uint4*)(qrow + h0);
            t.u4[1] = *(const uint4*)(qrow + h0 + 16);
            aQ[kk] = t.v;
        }
    }

    v8f acc[8];
#pragma unroll
    for (int i = 0; i < 8; ++i) acc[i] = (v8f){};
    float mrow[8], lrow[8];
#pragma unroll
    for (int r = 0; r < 8; ++r) { mrow[r] = -1e30f; lrow[r] = 0.0f; }

    const float scale = 0.08838834764831845f;   // 1/sqrt(128)
    const int numKT = (qt + 1) * 2;             // causal: keys <= qt*64+63

    // per-thread source/dest addressing for cooperative K/V tile staging
    const int vr = tid >> 2;                    // V row (key) 0..31
    const int vp = (tid & 3) * 32;              // V col chunk (32 bf16 = 64B)
    const __bf16* kbase = Ktb + ((size_t)b * H_ + tid) * T_;      // + kt*32
    const __bf16* vbase = Vb + ((size_t)b * T_ + vr) * H_ + vp;   // + kt*32*H_

    // ---- stage one 32-key tile (K^T 128x32 + V 32x128) into buffer `buf`
    auto stage_tile = [&](int kt, int buf) {
        const __bf16* ksrc = kbase + kt * 32;
        const __bf16* vsrc = vbase + (size_t)kt * 32 * H_;
#if HAS_ASYNC_LDS
#pragma unroll
        for (int j = 0; j < 4; ++j)
            async_copy16(&kbuf[buf][tid][j * 8], ksrc + j * 8);
#pragma unroll
        for (int j = 0; j < 4; ++j)
            async_copy16(&vbuf[buf][vr][vp + j * 8], vsrc + j * 8);
#else
#pragma unroll
        for (int j = 0; j < 4; ++j)
            *(uint4*)&kbuf[buf][tid][j * 8] = ((const uint4*)ksrc)[j];
#pragma unroll
        for (int j = 0; j < 4; ++j)
            *(uint4*)&vbuf[buf][vr][vp + j * 8] = ((const uint4*)vsrc)[j];
#endif
    };

    // prologue: fill buffer 0 with tile 0
    stage_tile(0, 0);
    wait_async0();
    __syncthreads();

    for (int kt = 0; kt < numKT; ++kt) {
        const int cur = kt & 1;

        // ---- kick off DMA for the next tile into the other buffer
        if (kt + 1 < numKT)
            stage_tile(kt + 1, cur ^ 1);

        // ---- scores: S (16x32) = Q (16x128) @ K^T (128x32), two 16-wide halves
        v8f c0 = (v8f){}, c1 = (v8f){};
#pragma unroll
        for (int kk = 0; kk < 4; ++kk) {
            Frag16 b0, b1;
#pragma unroll
            for (int s = 0; s < 16; ++s) {
                const int k = kk * 32 + (s & 7) + 8 * hf + 16 * (s >> 3);
                b0.v[s] = kbuf[cur][k][l16];
                b1.v[s] = kbuf[cur][k][16 + l16];
            }
            c0 = wmma_bf16(aQ[kk], b0.v, c0);
            c1 = wmma_bf16(aQ[kk], b1.v, c1);
        }

        // ---- online softmax (per query row; rows live on 16-lane groups)
        float alph[8];
        const int key0 = kt * 32 + l16;
#pragma unroll
        for (int r = 0; r < 8; ++r) {
            const int qIdx = qBase + r + 8 * hf;
            float s0 = c0[r] * scale; if (key0      > qIdx) s0 = -1e30f;
            float s1 = c1[r] * scale; if (key0 + 16 > qIdx) s1 = -1e30f;
            float mx = fmaxf(s0, s1);
            mx = fmaxf(mx, __shfl_xor(mx, 1));
            mx = fmaxf(mx, __shfl_xor(mx, 2));
            mx = fmaxf(mx, __shfl_xor(mx, 4));
            mx = fmaxf(mx, __shfl_xor(mx, 8));
            const float mnew = fmaxf(mrow[r], mx);
            alph[r] = __expf(mrow[r] - mnew);
            const float p0 = __expf(s0 - mnew);
            const float p1 = __expf(s1 - mnew);
            float ps = p0 + p1;
            ps += __shfl_xor(ps, 1);
            ps += __shfl_xor(ps, 2);
            ps += __shfl_xor(ps, 4);
            ps += __shfl_xor(ps, 8);
            lrow[r] = lrow[r] * alph[r] + ps;
            mrow[r] = mnew;
            pbuf[w][r + 8 * hf][l16]      = f2bf(p0);
            pbuf[w][r + 8 * hf][16 + l16] = f2bf(p1);
        }
#pragma unroll
        for (int hh = 0; hh < 8; ++hh)
#pragma unroll
            for (int r = 0; r < 8; ++r) acc[hh][r] *= alph[r];

        // ---- P re-layout (LDS, same wave: DS ops are in-order) -> A fragment
        Frag16 pa;
        {
            const __bf16* pr = &pbuf[w][l16][hf * 8];
            pa.u4[0] = *(const uint4*)(pr);
            pa.u4[1] = *(const uint4*)(pr + 16);
        }
        // ---- O += P (16x32) @ V (32x128)
#pragma unroll
        for (int hh = 0; hh < 8; ++hh) {
            Frag16 bv;
#pragma unroll
            for (int s = 0; s < 16; ++s)
                bv.v[s] = vbuf[cur][(s & 7) + 8 * hf + 16 * (s >> 3)][hh * 16 + l16];
            acc[hh] = wmma_bf16(pa.v, bv.v, acc[hh]);
        }

        // ---- my DMA for tile kt+1 done; all waves done reading tile kt
        wait_async0();
        __syncthreads();
    }

    // ---- epilogue: normalize by row sum, store fp32 (lanes coalesce over h)
#pragma unroll
    for (int r = 0; r < 8; ++r) {
        const float inv = 1.0f / lrow[r];
        const int m = qBase + r + 8 * hf;
#pragma unroll
        for (int hh = 0; hh < 8; ++hh)
            out[((size_t)b * T_ + m) * H_ + hh * 16 + l16] = acc[hh][r] * inv;
    }
}

// ---------------------------------------------------------------------------
extern "C" void kernel_launch(void* const* d_in, const int* in_sizes, int n_in,
                              void* d_out, int out_size, void* d_ws, size_t ws_size,
                              hipStream_t stream) {
    (void)in_sizes; (void)n_in; (void)out_size; (void)ws_size;
    const float* x  = (const float*)d_in[0];
    const float* Wk = (const float*)d_in[1];   // setup_inputs order: x, Wk, Wq, Wv
    const float* Wq = (const float*)d_in[2];
    const float* Wv = (const float*)d_in[3];
    float* out = (float*)d_out;

    const size_t QKV_BYTES = (size_t)B_ * T_ * H_ * sizeof(__bf16);  // 8 MB each
    char* ws = (char*)d_ws;
    __bf16* Qb  = (__bf16*)(ws);
    __bf16* Ktb = (__bf16*)(ws + QKV_BYTES);
    __bf16* Vb  = (__bf16*)(ws + 2 * QKV_BYTES);

    // Stage 1: 512 M-tiles x 3 matrices
    qkv_proj_kernel<<<dim3((B_ * T_) / 64, 3), 128, 0, stream>>>(
        x, Wq, Wk, Wv, Qb, Ktb, Vb);
    // Stage 2: 32 query tiles x 16 batches
    flash_attn_kernel<<<dim3(T_ / 64, B_), 128, 0, stream>>>(Qb, Ktb, Vb, out);
}
